// GlobalAttentionLayer_47974784696362
// MI455X (gfx1250) — matrix operand order repaired and checked by reading, not compiled, not run on hardware
//
#include <hip/hip_runtime.h>

typedef __attribute__((ext_vector_type(16))) _Float16 v16h;
typedef __attribute__((ext_vector_type(8)))  _Float16 v8h;
typedef __attribute__((ext_vector_type(8)))  float    v8f;

#define NUM_NODES  262144
#define NUM_GRAPHS 2048
#define D_X  256
#define D_G  128
#define D_IN 384
#define HID  256

union AFrag { v16h v; v8h h[2]; };
union BFrag { v16h v; uint4 u[2]; };
union Pack4 { _Float16 h[4]; uint2 u; };

// ---------------------------------------------------------------------------
// Prep: fold BN into per-column scale/bias; zero the segment-sum accumulator.
// ---------------------------------------------------------------------------
__global__ void prep_params_kernel(
    const float* __restrict__ b0,
    const float* __restrict__ g0, const float* __restrict__ be0,
    const float* __restrict__ m0, const float* __restrict__ v0,
    const float* __restrict__ b1,
    const float* __restrict__ g1, const float* __restrict__ be1,
    const float* __restrict__ m1, const float* __restrict__ v1,
    float* __restrict__ s0, float* __restrict__ t0,
    float* __restrict__ s1, float* __restrict__ t1,
    float* __restrict__ norm)
{
    int i = threadIdx.x;
    if (i < HID) {
        float sc0 = g0[i] * rsqrtf(v0[i] + 1e-3f);
        s0[i] = sc0;
        t0[i] = (b0[i] - m0[i]) * sc0 + be0[i];
        float sc1 = g1[i] * rsqrtf(v1[i] + 1e-3f);
        s1[i] = sc1;
        t1[i] = (b1[i] - m1[i]) * sc1 + be1[i];
    }
    for (int j = i; j < NUM_GRAPHS; j += blockDim.x) norm[j] = 0.0f;
}

// ---------------------------------------------------------------------------
// Pack W (K x 256, row-major f32) into f16 WMMA B-fragment order:
// fragment f = kt*16 + nt is a 32x16 tile; lane L owns 16 contiguous halfs
// = column nt*16 + (L&15), K = kt*32 + (L>>4)*8 + {0..7, 16..23}  (two B128 loads).
// ---------------------------------------------------------------------------
__global__ void pack_weights_kernel(const float* __restrict__ W,
                                    _Float16* __restrict__ Wp, int K)
{
    int frag = blockIdx.x * (blockDim.x >> 5) + (threadIdx.x >> 5);
    int lane = threadIdx.x & 31;
    int nfrags = (K / 32) * 16;
    if (frag >= nfrags) return;
    int kt = frag >> 4, nt = frag & 15;
    int col   = nt * 16 + (lane & 15);
    int kbase = kt * 32 + (lane >> 4) * 8;
    _Float16* dst = Wp + (size_t)frag * 512 + lane * 16;
#pragma unroll
    for (int k = 0; k < 16; ++k)
        dst[k] = (_Float16)W[(size_t)(kbase + (k < 8 ? k : k + 8)) * 256 + col];
}

// ---------------------------------------------------------------------------
// Fused MLP. 64 threads = 2 waves; 64 rows/block; each wave owns 32 rows
// (two 16-row M-tiles) so every B fragment load feeds two WMMAs, and all 8
// B fragments of an nt-half are loaded before the WMMA burst (1 wait per
// 16 wmma instead of 1 per wmma).
// ---------------------------------------------------------------------------
__global__ void __launch_bounds__(64)
attn_mlp_kernel(const float* __restrict__ x,
                const int*   __restrict__ batch_idx,
                const float* __restrict__ gfeat,
                const _Float16* __restrict__ W0p,
                const _Float16* __restrict__ W1p,
                const float* __restrict__ s0v, const float* __restrict__ t0v,
                const float* __restrict__ s1v, const float* __restrict__ t1v,
                const float* __restrict__ w2,  const float* __restrict__ b2,
                float* __restrict__ out, float* __restrict__ norm)
{
    __shared__ _Float16 Atile[64 * D_IN];   // 48 KB
    __shared__ _Float16 H1s[64 * HID];      // 32 KB
    __shared__ float    rowbuf[64];
    __shared__ int      bidx_s[64];

    const int tid = threadIdx.x;            // 0..63
    const int rowBase = blockIdx.x * 64;

    bidx_s[tid] = batch_idx[rowBase + tid];
    __syncthreads();

    // Stage concat(x, gf[batch_idx]) as f16 into LDS: b128 global reads,
    // packed 8-byte LDS writes. This is the HBM-bound 256 MB stream.
    for (int idx = tid; idx < 64 * (D_IN / 4); idx += 64) {
        int r = idx / (D_IN / 4);
        int c = (idx % (D_IN / 4)) * 4;
        float4 vv;
        if (c < D_X)
            vv = *(const float4*)(x + (size_t)(rowBase + r) * D_X + c);
        else
            vv = *(const float4*)(gfeat + (size_t)bidx_s[r] * D_G + (c - D_X));
        Pack4 pk;
        pk.h[0] = (_Float16)vv.x;  pk.h[1] = (_Float16)vv.y;
        pk.h[2] = (_Float16)vv.z;  pk.h[3] = (_Float16)vv.w;
        *(uint2*)&Atile[r * D_IN + c] = pk.u;
    }
    __syncthreads();

    const int wave    = tid >> 5;
    const int lane    = tid & 31;
    const int laneCol = lane & 15;
    const int half16  = lane >> 4;
    const v8f vzero = {0.f, 0.f, 0.f, 0.f, 0.f, 0.f, 0.f, 0.f};

    // A-fragment bases for the wave's two 16-row M-tiles.
    const _Float16* A0 = &Atile[(wave * 32 + laneCol) * D_IN];
    const _Float16* A1 = A0 + 16 * D_IN;

    // ---------------- Layer 0: [32 x 384] @ [384 x 256] ----------------
    for (int nh = 0; nh < 2; ++nh) {
        v8f acc0[8], acc1[8];
#pragma unroll
        for (int j = 0; j < 8; ++j) { acc0[j] = vzero; acc1[j] = vzero; }
        for (int kt = 0; kt < D_IN / 32; ++kt) {
            AFrag a0, a1;
            const _Float16* ap0 = A0 + kt * 32 + half16 * 8;
            a0.h[0] = *(const v8h*)(ap0);
            a0.h[1] = *(const v8h*)(ap0 + 16);
            const _Float16* ap1 = A1 + kt * 32 + half16 * 8;
            a1.h[0] = *(const v8h*)(ap1);
            a1.h[1] = *(const v8h*)(ap1 + 16);
            // Batch all 8 B fragments (16 x global_load_b128, const offsets).
            BFrag b[8];
            const _Float16* wp = W0p + (size_t)(kt * 16 + nh * 8) * 512 + lane * 16;
#pragma unroll
            for (int j = 0; j < 8; ++j) {
                b[j].u[0] = *(const uint4*)(wp + j * 512);
                b[j].u[1] = *(const uint4*)(wp + j * 512 + 8);
            }
#pragma unroll
            for (int j = 0; j < 8; ++j) {
                acc0[j] = __builtin_amdgcn_wmma_f32_16x16x32_f16(
                    false, a0.v, false, b[j].v, (short)0, acc0[j], false, false);
                acc1[j] = __builtin_amdgcn_wmma_f32_16x16x32_f16(
                    false, a1.v, false, b[j].v, (short)0, acc1[j], false, false);
            }
        }
        // BN + ReLU, C-layout (lane = col, VGPR r = row) -> LDS f16
#pragma unroll
        for (int j = 0; j < 8; ++j) {
            int col = (nh * 8 + j) * 16 + laneCol;
            float sc = s0v[col], tb = t0v[col];
#pragma unroll
            for (int r = 0; r < 8; ++r) {
                int m = wave * 32 + half16 * 8 + r;
                float v0x = acc0[j][r] * sc + tb;
                v0x = v0x > 0.f ? v0x : 0.f;
                H1s[m * HID + col] = (_Float16)v0x;
                float v1x = acc1[j][r] * sc + tb;
                v1x = v1x > 0.f ? v1x : 0.f;
                H1s[(m + 16) * HID + col] = (_Float16)v1x;
            }
        }
    }
    __syncthreads();

    // ------- Layer 1: [32 x 256] @ [256 x 256] + fused dot with w2 -------
    float part0[8], part1[8];
#pragma unroll
    for (int r = 0; r < 8; ++r) { part0[r] = 0.f; part1[r] = 0.f; }

    const _Float16* H0 = &H1s[(wave * 32 + laneCol) * HID];
    const _Float16* H1b = H0 + 16 * HID;
    for (int nh = 0; nh < 2; ++nh) {
        v8f acc0[8], acc1[8];
#pragma unroll
        for (int j = 0; j < 8; ++j) { acc0[j] = vzero; acc1[j] = vzero; }
        for (int kt = 0; kt < HID / 32; ++kt) {
            AFrag a0, a1;
            const _Float16* ap0 = H0 + kt * 32 + half16 * 8;
            a0.h[0] = *(const v8h*)(ap0);
            a0.h[1] = *(const v8h*)(ap0 + 16);
            const _Float16* ap1 = H1b + kt * 32 + half16 * 8;
            a1.h[0] = *(const v8h*)(ap1);
            a1.h[1] = *(const v8h*)(ap1 + 16);
            BFrag b[8];
            const _Float16* wp = W1p + (size_t)(kt * 16 + nh * 8) * 512 + lane * 16;
#pragma unroll
            for (int j = 0; j < 8; ++j) {
                b[j].u[0] = *(const uint4*)(wp + j * 512);
                b[j].u[1] = *(const uint4*)(wp + j * 512 + 8);
            }
#pragma unroll
            for (int j = 0; j < 8; ++j) {
                acc0[j] = __builtin_amdgcn_wmma_f32_16x16x32_f16(
                    false, a0.v, false, b[j].v, (short)0, acc0[j], false, false);
                acc1[j] = __builtin_amdgcn_wmma_f32_16x16x32_f16(
                    false, a1.v, false, b[j].v, (short)0, acc1[j], false, false);
            }
        }
#pragma unroll
        for (int j = 0; j < 8; ++j) {
            int col = (nh * 8 + j) * 16 + laneCol;
            float sc = s1v[col], tb = t1v[col], wv = w2[col];
#pragma unroll
            for (int r = 0; r < 8; ++r) {
                float v0x = acc0[j][r] * sc + tb;
                v0x = v0x > 0.f ? v0x : 0.f;
                part0[r] += v0x * wv;              // fused Dense(1)
                float v1x = acc1[j][r] * sc + tb;
                v1x = v1x > 0.f ? v1x : 0.f;
                part1[r] += v1x * wv;
            }
        }
    }

    // Reduce the 256-wide dot across the 16 lanes of each half-wave.
#pragma unroll
    for (int m = 1; m < 16; m <<= 1) {
#pragma unroll
        for (int r = 0; r < 8; ++r) {
            part0[r] += __shfl_xor(part0[r], m, 32);
            part1[r] += __shfl_xor(part1[r], m, 32);
        }
    }
    if (laneCol == 0) {
#pragma unroll
        for (int r = 0; r < 8; ++r) {
            rowbuf[wave * 32 + half16 * 8 + r]      = part0[r];
            rowbuf[wave * 32 + half16 * 8 + r + 16] = part1[r];
        }
    }
    __syncthreads();

    // exp, emit att, pre-reduce segment sum within the block (sorted idx).
    float att = expf(rowbuf[tid] + b2[0]);
    out[rowBase + tid] = att;
    __syncthreads();
    rowbuf[tid] = att;
    __syncthreads();
    {
        int b = bidx_s[tid];
        if (tid == 0 || bidx_s[tid - 1] != b) {    // run leader
            float s = 0.f;
            for (int k = tid; k < 64 && bidx_s[k] == b; ++k) s += rowbuf[k];
            atomicAdd(&norm[b], s);
        }
    }
}

// ---------------------------------------------------------------------------
__global__ void normalize_kernel(float* __restrict__ out,
                                 const float* __restrict__ norm,
                                 const int* __restrict__ bidx, int n)
{
    int i = blockIdx.x * blockDim.x + threadIdx.x;
    if (i < n) out[i] = out[i] / norm[bidx[i]];
}

// ---------------------------------------------------------------------------
extern "C" void kernel_launch(void* const* d_in, const int* in_sizes, int n_in,
                              void* d_out, int out_size, void* d_ws, size_t ws_size,
                              hipStream_t stream)
{
    const float* x         = (const float*)d_in[0];
    const int*   batch_idx = (const int*)  d_in[1];
    const float* gfeat     = (const float*)d_in[2];
    const float* W0        = (const float*)d_in[3];
    const float* b0        = (const float*)d_in[4];
    const float* W1        = (const float*)d_in[5];
    const float* b1        = (const float*)d_in[6];
    const float* W2        = (const float*)d_in[7];
    const float* b2        = (const float*)d_in[8];
    const float* gamma0    = (const float*)d_in[9];
    const float* beta0     = (const float*)d_in[10];
    const float* mean0     = (const float*)d_in[11];
    const float* var0      = (const float*)d_in[12];
    const float* gamma1    = (const float*)d_in[13];
    const float* beta1     = (const float*)d_in[14];
    const float* mean1     = (const float*)d_in[15];
    const float* var1      = (const float*)d_in[16];
    float* out = (float*)d_out;

    char* ws = (char*)d_ws;
    float*     s0   = (float*)(ws + 0);
    float*     t0   = (float*)(ws + 1024);
    float*     s1   = (float*)(ws + 2048);
    float*     t1   = (float*)(ws + 3072);
    float*     norm = (float*)(ws + 4096);               // 2048 f32
    _Float16*  W0p  = (_Float16*)(ws + 12288);           // 384*256 f16 = 192 KB
    _Float16*  W1p  = (_Float16*)(ws + 208896);          // 256*256 f16 = 128 KB

    prep_params_kernel<<<1, 256, 0, stream>>>(
        b0, gamma0, beta0, mean0, var0,
        b1, gamma1, beta1, mean1, var1,
        s0, t0, s1, t1, norm);

    pack_weights_kernel<<<48, 128, 0, stream>>>(W0, W0p, D_IN); // 192 frags
    pack_weights_kernel<<<32, 128, 0, stream>>>(W1, W1p, HID);  // 128 frags

    attn_mlp_kernel<<<NUM_NODES / 64, 64, 0, stream>>>(
        x, batch_idx, gfeat, W0p, W1p, s0, t0, s1, t1, W2, b2, out, norm);

    normalize_kernel<<<(NUM_NODES + 255) / 256, 256, 0, stream>>>(
        out, norm, batch_idx, NUM_NODES);
}
